// MoEReduceRSTensorParallel_36816459661328
// MI455X (gfx1250) — matrix operand order repaired and checked by reading, not compiled, not run on hardware
//
#include <hip/hip_runtime.h>

typedef __attribute__((ext_vector_type(16))) _Float16 v16h;
typedef __attribute__((ext_vector_type(8)))  _Float16 v8h;
typedef __attribute__((ext_vector_type(4)))  _Float16 v4h;
typedef __attribute__((ext_vector_type(8)))  float    v8f;

// Problem constants (match reference)
constexpr int NTOK = 8192;
constexpr int TOPK = 2;
constexpr int EC   = 8;      // experts
constexpr int KC   = 512;    // intermediate (GEMM K)
constexpr int HC   = 2048;   // hidden (GEMM N)
constexpr int TC   = NTOK * TOPK;        // 16384 rows
constexpr int MT   = 64;                 // rows per supertile (4 x 16)
constexpr int PADT = TC + EC * MT;       // per-expert 64-padding worst case (16896)
constexpr int NTILES = PADT / MT;        // 264 supertiles
constexpr int INVALID = 0xFFFFF;

// ---------------- zero output ----------------
__global__ void zero_f32(float* __restrict__ p, int n) {
    int i = (blockIdx.x * blockDim.x + threadIdx.x) * 4;
    if (i < n) *(float4*)(p + i) = make_float4(0.f, 0.f, 0.f, 0.f);
}

// ---------------- f32 -> f16 convert (linear, for A) ----------------
__global__ void cvt_f32_f16(const float* __restrict__ src, _Float16* __restrict__ dst, int n) {
    int i = (blockIdx.x * blockDim.x + threadIdx.x) * 4;
    if (i < n) {
        float4 v = *(const float4*)(src + i);
        v4h h;
        h.x = (_Float16)v.x; h.y = (_Float16)v.y;
        h.z = (_Float16)v.z; h.w = (_Float16)v.w;
        *(v4h*)(dst + i) = h;
    }
}

// ---------------- f32 -> f16 convert + transpose (for W) ----------------
// src: W [E][K][H] f32 -> dst: WT [E][H][K] f16 ; 32x32 tiles via LDS.
__global__ __launch_bounds__(256) void cvt_transpose_w(const float* __restrict__ src,
                                                       _Float16* __restrict__ dst) {
    __shared__ _Float16 tile[32][33];        // +1 pad: avoid bank conflicts
    const int e  = blockIdx.z;
    const int n0 = blockIdx.x * 32;          // H tile base
    const int k0 = blockIdx.y * 32;          // K tile base
    const int c  = threadIdx.x & 31;
    const int r  = threadIdx.x >> 5;         // 0..7
    #pragma unroll
    for (int j = 0; j < 4; ++j) {
        int k = k0 + r + j * 8;
        tile[r + j * 8][c] = (_Float16)src[((size_t)e * KC + k) * HC + n0 + c];
    }
    __syncthreads();
    #pragma unroll
    for (int j = 0; j < 4; ++j) {
        int n = n0 + r + j * 8;
        dst[((size_t)e * HC + n) * KC + k0 + c] = tile[c][r + j * 8];
    }
}

// ---------------- expert-sorted, 64-padded row list ----------------
// entry = (expert << 20) | row ; row == INVALID marks padding.
// Valid rows packed first per segment -> supertile all-invalid iff entry0 is,
// and every 64-row supertile is single-expert.
__global__ void build_row_list(const int* __restrict__ ids, int* __restrict__ list) {
    __shared__ int cnt[EC], segBase[EC], segEnd[EC], fill[EC];
    const int tid = threadIdx.x;
    if (tid < EC) cnt[tid] = 0;
    __syncthreads();
    for (int r = tid; r < TC; r += blockDim.x) atomicAdd(&cnt[ids[r]], 1);
    __syncthreads();
    if (tid == 0) {
        int off = 0;
        for (int e = 0; e < EC; ++e) {
            segBase[e] = off; fill[e] = 0;
            off += (cnt[e] + (MT - 1)) & ~(MT - 1);
            segEnd[e] = off;
        }
    }
    __syncthreads();
    for (int p = tid; p < PADT; p += blockDim.x) {
        int e = 0;
        for (int q = 0; q < EC; ++q)
            if (p >= segBase[q] && p < segEnd[q]) e = q;
        list[p] = (e << 20) | INVALID;
    }
    __syncthreads();
    for (int r = tid; r < TC; r += blockDim.x) {
        int e = ids[r];
        int pos = segBase[e] + atomicAdd(&fill[e], 1);
        list[pos] = (e << 20) | r;
    }
}

// ---------------- grouped GEMM + weighted scatter ----------------
// grid.x = 64-row supertile (single expert), grid.y = 128-column block.
// 256 threads = 8 waves; wave -> 16 columns, 4 accumulators (4 M-subtiles).
// One B fragment feeds 4 independent WMMAs (4x B reuse, no RAW hazards).
__global__ __launch_bounds__(256) void moe_down_gemm(
    const _Float16* __restrict__ A16,   // [TC][KC]
    const _Float16* __restrict__ W16T,  // [EC][HC][KC] (transposed)
    const int*      __restrict__ list,  // [PADT]
    const float*    __restrict__ topkw, // [TC] flat
    float*          __restrict__ out)   // [NTOK][HC]
{
    __shared__ _Float16 Atile[MT][KC];  // 64 KB
    __shared__ int   ent[MT];
    __shared__ float rw[MT];
    __shared__ int   tok[MT];

    const int tid = threadIdx.x;

    if (tid < MT) {
        int e = list[blockIdx.x * MT + tid];
        ent[tid] = e;
        int row = e & 0xFFFFF;
        // padding rows: weight 0 -> branchless "+= 0.0f" into out[0][n]
        if (row != INVALID) { rw[tid] = topkw[row]; tok[tid] = row >> 1; }
        else                { rw[tid] = 0.0f;       tok[tid] = 0; }
    }
    __syncthreads();

    if ((ent[0] & 0xFFFFF) == INVALID) return;   // block-uniform: tile fully padding

    // Gather A supertile into LDS: 4 threads per row, 128 halfs (256 B) each.
    {
        int m = tid >> 2;               // 0..63
        int q = tid & 3;                // 0..3
        int kbase = q * 128;
        int row = ent[m] & 0xFFFFF;
        uint4* dstp = (uint4*)&Atile[m][kbase];
        if (row != INVALID) {
            const uint4* srcp = (const uint4*)(A16 + (size_t)row * KC + kbase);
            #pragma unroll
            for (int j = 0; j < 16; ++j) dstp[j] = srcp[j];
        } else {
            #pragma unroll
            for (int j = 0; j < 16; ++j) dstp[j] = make_uint4(0u, 0u, 0u, 0u);
        }
    }
    __syncthreads();

    const int wave = tid >> 5, lane = tid & 31;
    const int hx = lane >> 4, lm = lane & 15;           // lane half, lane-in-half
    const int n = blockIdx.y * 128 + wave * 16 + lm;    // global output column
    const int e0 = ent[0] >> 20;                        // tile's expert
    const _Float16* WbT = W16T + ((size_t)e0 * HC + n) * KC;  // this lane's B row

    v8f c[4];
    #pragma unroll
    for (int s = 0; s < 4; ++s) c[s] = v8f{0.f, 0.f, 0.f, 0.f, 0.f, 0.f, 0.f, 0.f};

    #pragma unroll 2
    for (int kb = 0; kb < KC; kb += 32) {
        // B 32x16 f16 frag: lane=N; K = hx*16 + i -> contiguous 32 B in WT
        v16h b = *(const v16h*)(WbT + kb + hx * 16);
        #pragma unroll
        for (int s = 0; s < 4; ++s) {
            // A 16x32 f16 frag: lane=M; half0 K=0..15, half1 K=16..31
            const _Float16* ar = &Atile[s * 16 + lm][kb + hx * 16];
            v8h alo = *(const v8h*)ar;
            v8h ahi = *(const v8h*)(ar + 8);
            v16h a = __builtin_shufflevector(alo, ahi,
                                             0, 1, 2, 3, 4, 5, 6, 7,
                                             8, 9, 10, 11, 12, 13, 14, 15);
            c[s] = __builtin_amdgcn_wmma_f32_16x16x32_f16(
                       false, a, false, b, (short)0, c[s], false, false);
        }
    }

    // D layout: VGPR j -> M = j + hx*8, N = lm. Branchless weighted scatter.
    #pragma unroll
    for (int s = 0; s < 4; ++s) {
        #pragma unroll
        for (int j = 0; j < 8; ++j) {
            int m = s * 16 + j + hx * 8;
            atomicAdd(&out[(size_t)tok[m] * HC + n], c[s][j] * rw[m]);
        }
    }
}

extern "C" void kernel_launch(void* const* d_in, const int* in_sizes, int n_in,
                              void* d_out, int out_size, void* d_ws, size_t ws_size,
                              hipStream_t stream) {
    (void)in_sizes; (void)n_in; (void)out_size; (void)ws_size;
    const float* inter = (const float*)d_in[0];   // [TC][KC] f32
    const float* dw    = (const float*)d_in[1];   // [EC][KC][HC] f32
    const int*   ids   = (const int*)d_in[2];     // [NTOK][TOPK] int32 (flat per-row expert)
    const float* tw    = (const float*)d_in[3];   // [NTOK][TOPK] f32
    float* out = (float*)d_out;

    char* ws = (char*)d_ws;
    _Float16* A16  = (_Float16*)ws;
    _Float16* W16T = (_Float16*)(ws + (size_t)TC * KC * sizeof(_Float16));
    int* list = (int*)(ws + (size_t)TC * KC * sizeof(_Float16)
                          + (size_t)EC * KC * HC * sizeof(_Float16));

    const int nOut = NTOK * HC;
    zero_f32<<<nOut / 4 / 256, 256, 0, stream>>>(out, nOut);

    const int nA = TC * KC;
    cvt_f32_f16<<<(nA / 4 + 255) / 256, 256, 0, stream>>>(inter, A16, nA);

    dim3 tgrid(HC / 32, KC / 32, EC);
    cvt_transpose_w<<<tgrid, 256, 0, stream>>>(dw, W16T);

    build_row_list<<<1, 256, 0, stream>>>(ids, list);

    dim3 grid(NTILES, HC / 128);
    moe_down_gemm<<<grid, 256, 0, stream>>>(A16, W16T, list, tw, out);
}
